// MultiHeadAttention_28741921145156
// MI455X (gfx1250) — compile-verified, matrix operand order checked by reading, not run on hardware
//
#include <hip/hip_runtime.h>

// Problem constants from the reference
#define S_LEN 2048
#define DIMSZ 2048
#define NBATCH 2
#define NHEAD 16
#define HDIM 128

typedef __attribute__((ext_vector_type(16))) __bf16 v16bf;
typedef __attribute__((ext_vector_type(8)))  float  v8f;

union FragBF { uint4 u[2]; v16bf v; };

__device__ __forceinline__ unsigned short f2bf(float f) {
  unsigned int u = __float_as_uint(f);
  u += 0x7FFFu + ((u >> 16) & 1u);   // round-to-nearest-even
  return (unsigned short)(u >> 16);
}
__device__ __forceinline__ float bf2f(unsigned short h) {
  return __uint_as_float((unsigned int)h << 16);
}
__device__ __forceinline__ unsigned int pack2(float a, float b) {
  return (unsigned int)f2bf(a) | ((unsigned int)f2bf(b) << 16);
}
__device__ __forceinline__ v8f zero8() {
  v8f z;
#pragma unroll
  for (int i = 0; i < 8; ++i) z[i] = 0.0f;
  return z;
}
__device__ __forceinline__ v8f wmma_bf16(v16bf a, v16bf b, v8f c) {
  return __builtin_amdgcn_wmma_f32_16x16x32_bf16(false, a, false, b,
                                                 (short)0, c, false, false);
}

// CDNA5 async global->LDS copy (ASYNCcnt-tracked), 16B per lane.
__device__ __forceinline__ void async_b128(unsigned lds_off, const void* gaddr) {
  asm volatile("global_load_async_to_lds_b128 %0, %1, off"
               :: "v"(lds_off), "v"(gaddr) : "memory");
}
__device__ __forceinline__ void wait_async0() {
  asm volatile("s_wait_asynccnt 0" ::: "memory");
}
__device__ __forceinline__ unsigned lds_off_of(const void* p) {
  return (unsigned)(unsigned long long)p;  // low 32 bits = LDS byte offset
}

// ---------------------------------------------------------------------------
// One-shot fp32 -> bf16 conversion (x and the four weight matrices).
// ---------------------------------------------------------------------------
__global__ __launch_bounds__(256)
void to_bf16(const float* __restrict__ src, unsigned short* __restrict__ dst) {
  const size_t i = ((size_t)blockIdx.x * 256 + threadIdx.x) * 4;
  const float4 f = *(const float4*)(src + i);
  *(uint2*)(dst + i) = make_uint2(pack2(f.x, f.y), pack2(f.z, f.w));
}

// ---------------------------------------------------------------------------
// GEMM (NT): C[M,N] = A[M,K] * W[N,K]^T, bf16 in, bf16 or fp32 out.
// 128x128 block tile, 8 waves of 32x64, K step 32, double-buffered LDS
// filled with global_load_async_to_lds_b128 overlapping the WMMAs.
// ---------------------------------------------------------------------------
#define LDT 40  // padded LDS k-stride in ushorts (80 B = 5 x 16 B)

template <bool OUT_BF16>
__global__ __launch_bounds__(256)
void gemm_nt_bf16(const unsigned short* __restrict__ A,
                  const unsigned short* __restrict__ W,
                  void* __restrict__ Cout, int K, int N) {
  __shared__ unsigned short lA[2][128 * LDT];
  __shared__ unsigned short lB[2][128 * LDT];

  const int tid  = threadIdx.x;
  const int lane = tid & 31;
  const int wid  = tid >> 5;
  const int wm   = wid & 3;          // wave row 0..3  (32 rows each)
  const int wn   = wid >> 2;         // wave col 0..1  (64 cols each)
  const int bm   = blockIdx.y * 128;
  const int bn   = blockIdx.x * 128;
  const int half = lane >> 4;
  const int l15  = lane & 15;
  const int kb   = half * 8;         // per-lane K sub-offset for 16-bit frags

  v8f acc[2][4];
#pragma unroll
  for (int mt = 0; mt < 2; ++mt)
#pragma unroll
    for (int nt = 0; nt < 4; ++nt) acc[mt][nt] = zero8();

  const int crow = tid >> 2;         // 0..63
  const int cchk = (tid & 3) * 16;   // 16B chunk within a 64B row

  auto issue_tile = [&](int buf, int k0) {
#pragma unroll
    for (int p = 0; p < 2; ++p) {
      const int row = crow + p * 64;
      async_b128(lds_off_of(&lA[buf][row * LDT]) + cchk,
                 (const char*)(A + (size_t)(bm + row) * K + k0) + cchk);
      async_b128(lds_off_of(&lB[buf][row * LDT]) + cchk,
                 (const char*)(W + (size_t)(bn + row) * K + k0) + cchk);
    }
  };

  issue_tile(0, 0);
  int buf = 0;
  for (int k0 = 0; k0 < K; k0 += 32, buf ^= 1) {
    wait_async0();
    __syncthreads();                     // tile `buf` ready; `buf^1` free
    if (k0 + 32 < K) issue_tile(buf ^ 1, k0 + 32);

    FragBF af[2], bf[4];
#pragma unroll
    for (int mt = 0; mt < 2; ++mt) {
      const unsigned short* p = &lA[buf][(wm * 32 + mt * 16 + l15) * LDT + kb];
      af[mt].u[0] = *(const uint4*)p;
      af[mt].u[1] = *(const uint4*)(p + 16);
    }
#pragma unroll
    for (int nt = 0; nt < 4; ++nt) {
      const unsigned short* p = &lB[buf][(wn * 64 + nt * 16 + l15) * LDT + kb];
      bf[nt].u[0] = *(const uint4*)p;
      bf[nt].u[1] = *(const uint4*)(p + 16);
    }
#pragma unroll
    for (int mt = 0; mt < 2; ++mt)
#pragma unroll
      for (int nt = 0; nt < 4; ++nt)
        acc[mt][nt] = wmma_bf16(af[mt].v, bf[nt].v, acc[mt][nt]);
  }

#pragma unroll
  for (int mt = 0; mt < 2; ++mt)
#pragma unroll
    for (int nt = 0; nt < 4; ++nt)
#pragma unroll
      for (int r = 0; r < 8; ++r) {
        const int row = bm + wm * 32 + mt * 16 + r + half * 8;
        const int col = bn + wn * 64 + nt * 16 + l15;
        if (OUT_BF16)
          ((unsigned short*)Cout)[(size_t)row * N + col] = f2bf(acc[mt][nt][r]);
        else
          ((float*)Cout)[(size_t)row * N + col] = acc[mt][nt][r];
      }
}

// ---------------------------------------------------------------------------
// Rotate-half RoPE applied in place to bf16 Q and K.
// ---------------------------------------------------------------------------
__global__ __launch_bounds__(256)
void rope_qk(unsigned short* __restrict__ q, unsigned short* __restrict__ k) {
  const int idx = blockIdx.x * 256 + threadIdx.x;  // B*S*H*64 threads
  const int d = idx & 63;
  const int h = (idx >> 6) & (NHEAD - 1);
  const int s = (idx >> 10) & (S_LEN - 1);
  const int b = idx >> 21;
  const float inv = __expf(-((float)(2 * d) / (float)HDIM) * 9.210340371976184f);
  const float ang = (float)s * inv;
  float sn, cs;
  __sincosf(ang, &sn, &cs);
  const size_t base = ((size_t)(b * S_LEN + s)) * DIMSZ + h * HDIM + d;
  const float q1 = bf2f(q[base]), q2 = bf2f(q[base + 64]);
  q[base]      = f2bf(q1 * cs - q2 * sn);
  q[base + 64] = f2bf(q2 * cs + q1 * sn);
  const float k1 = bf2f(k[base]), k2 = bf2f(k[base + 64]);
  k[base]      = f2bf(k1 * cs - k2 * sn);
  k[base + 64] = f2bf(k2 * cs + k1 * sn);
}

// ---------------------------------------------------------------------------
// Causal flash attention over bf16 Q/K/V, bf16 out. Block = 128 q rows x
// (head, batch); 8 waves x 16 q rows. Key chunk = 32. K tile arrives via
// async global->LDS copy; V is transposed through registers into [d][key].
// ---------------------------------------------------------------------------
__global__ __launch_bounds__(256)
void flash_attn(const unsigned short* __restrict__ q,
                const unsigned short* __restrict__ k,
                const unsigned short* __restrict__ v,
                unsigned short* __restrict__ o) {
  __shared__ unsigned short lK[32 * 136];    // K chunk [key][d], 272B stride
  __shared__ unsigned short lVt[128 * 40];   // V chunk transposed [d][key]
  __shared__ unsigned short lP[8 * 16 * 32]; // per-wave probability tiles

  const int tid  = threadIdx.x;
  const int lane = tid & 31;
  const int w    = tid >> 5;
  const int half = lane >> 4;
  const int l15  = lane & 15;
  const int kb   = half * 8;
  const int qblk = blockIdx.x;
  const int h    = blockIdx.y;
  const int b    = blockIdx.z;
  const int qbase = qblk * 128 + w * 16;
  const float sm_scale = 0.08838834764831845f;  // 128^-0.5
  const float NEG = -3.0e38f;

  // Q fragments (4 x 16x32 bf16) straight from global (post-RoPE, bf16)
  FragBF qf[4];
  {
    const unsigned short* qp =
        q + ((size_t)(b * S_LEN + qbase + l15)) * DIMSZ + h * HDIM;
#pragma unroll
    for (int c = 0; c < 4; ++c) {
      qf[c].u[0] = *(const uint4*)(qp + c * 32 + kb);
      qf[c].u[1] = *(const uint4*)(qp + c * 32 + kb + 16);
    }
  }

  float mstate[8], lstate[8];
  v8f acc[8];
#pragma unroll
  for (int r = 0; r < 8; ++r) { mstate[r] = NEG; lstate[r] = 0.0f; }
#pragma unroll
  for (int t = 0; t < 8; ++t) acc[t] = zero8();

  const int kend = (qblk + 1) * 128;  // causal: block-uniform key bound
  for (int kc = 0; kc < kend; kc += 32) {
    __syncthreads();
    // K chunk: async copy, 32 rows x 256B = 512 x 16B chunks
#pragma unroll
    for (int p = 0; p < 2; ++p) {
      const int idx = p * 256 + tid;
      const int rl = idx >> 4;          // key row 0..31
      const int ch = (idx & 15) * 16;   // 16B chunk within 256B row
      const char* ga =
          (const char*)(k + ((size_t)(b * S_LEN + kc + rl)) * DIMSZ + h * HDIM) + ch;
      async_b128(lds_off_of(&lK[rl * 136]) + ch, ga);
    }
    // V chunk: registers -> transposed LDS [d][key]
#pragma unroll
    for (int p = 0; p < 2; ++p) {
      const int idx = p * 256 + tid;
      const int rl = idx >> 4;          // key row 0..31
      const int dq = (idx & 15) * 8;    // d offset 0..120
      const uint4 vv = *(const uint4*)(
          v + ((size_t)(b * S_LEN + kc + rl)) * DIMSZ + h * HDIM + dq);
      const unsigned short* e = (const unsigned short*)&vv;
#pragma unroll
      for (int j = 0; j < 8; ++j) lVt[(dq + j) * 40 + rl] = e[j];
    }
    wait_async0();
    __syncthreads();

    // S = Q * K^T  (16 x 32 scores, two 16x16 accumulators)
    v8f sc[2] = { zero8(), zero8() };
#pragma unroll
    for (int n = 0; n < 2; ++n)
#pragma unroll
      for (int c = 0; c < 4; ++c) {
        FragBF kf;
        const unsigned short* p = &lK[(n * 16 + l15) * 136 + c * 32 + kb];
        kf.u[0] = *(const uint4*)p;
        kf.u[1] = *(const uint4*)(p + 16);
        sc[n] = wmma_bf16(qf[c].v, kf.v, sc[n]);
      }

    // Online softmax (rows split over lane halves; 16 lanes hold a row's cols)
    float scl[8];
    unsigned short* pw = &lP[w * 512];
#pragma unroll
    for (int r = 0; r < 8; ++r) {
      const int qi = qbase + r + half * 8;
      float s0 = (kc + l15 <= qi)      ? sc[0][r] * sm_scale : NEG;
      float s1 = (kc + 16 + l15 <= qi) ? sc[1][r] * sm_scale : NEG;
      float mx = fmaxf(s0, s1);
#pragma unroll
      for (int m = 1; m <= 8; m <<= 1) mx = fmaxf(mx, __shfl_xor(mx, m, 32));
      const float mnew = fmaxf(mstate[r], mx);
      const float resc = __expf(mstate[r] - mnew);
      const float p0 = __expf(s0 - mnew);
      const float p1 = __expf(s1 - mnew);
      float rs = p0 + p1;
#pragma unroll
      for (int m = 1; m <= 8; m <<= 1) rs += __shfl_xor(rs, m, 32);
      lstate[r] = lstate[r] * resc + rs;
      mstate[r] = mnew;
      scl[r] = resc;
      const int row = r + half * 8;
      pw[row * 32 + l15]      = f2bf(p0);
      pw[row * 32 + 16 + l15] = f2bf(p1);
    }
#pragma unroll
    for (int t = 0; t < 8; ++t)
#pragma unroll
      for (int r = 0; r < 8; ++r) acc[t][r] *= scl[r];

    __builtin_amdgcn_wave_barrier();  // order same-wave LDS store -> load of P

    FragBF pf;
    {
      const unsigned short* pp = &pw[l15 * 32 + kb];
      pf.u[0] = *(const uint4*)pp;
      pf.u[1] = *(const uint4*)(pp + 16);
    }
    // O += P * V   (8 output column tiles of 16)
#pragma unroll
    for (int t = 0; t < 8; ++t) {
      FragBF vf;
      const unsigned short* vp = &lVt[(t * 16 + l15) * 40 + kb];
      vf.u[0] = *(const uint4*)vp;
      vf.u[1] = *(const uint4*)(vp + 16);
      acc[t] = wmma_bf16(pf.v, vf.v, acc[t]);
    }
  }

#pragma unroll
  for (int r = 0; r < 8; ++r) {
    const float rinv = 1.0f / lstate[r];
    const int row = qbase + r + half * 8;
    const size_t base = ((size_t)(b * S_LEN + row)) * DIMSZ + h * HDIM + l15;
#pragma unroll
    for (int t = 0; t < 8; ++t) o[base + t * 16] = f2bf(acc[t][r] * rinv);
  }
}

// ---------------------------------------------------------------------------
extern "C" void kernel_launch(void* const* d_in, const int* in_sizes, int n_in,
                              void* d_out, int out_size, void* d_ws, size_t ws_size,
                              hipStream_t stream) {
  (void)in_sizes; (void)n_in; (void)out_size; (void)ws_size;
  const float* x  = (const float*)d_in[0];
  // d_in[1] = causal mask (implicit in flash_attn)
  const float* wq = (const float*)d_in[2];
  const float* wk = (const float*)d_in[3];
  const float* wv = (const float*)d_in[4];
  const float* wo = (const float*)d_in[5];

  const size_t actE = (size_t)NBATCH * S_LEN * DIMSZ;  // 8,388,608
  const size_t wE   = (size_t)DIMSZ * DIMSZ;           // 4,194,304

  unsigned short* xb  = (unsigned short*)d_ws;
  unsigned short* wqb = xb + actE;
  unsigned short* wkb = wqb + wE;
  unsigned short* wvb = wkb + wE;
  unsigned short* wob = wvb + wE;
  unsigned short* qb  = wob + wE;
  unsigned short* kbf = qb + actE;
  unsigned short* vbf = kbf + actE;
  unsigned short* abf = vbf + actE;

  to_bf16<<<actE / 1024, 256, 0, stream>>>(x, xb);
  to_bf16<<<wE / 1024, 256, 0, stream>>>(wq, wqb);
  to_bf16<<<wE / 1024, 256, 0, stream>>>(wk, wkb);
  to_bf16<<<wE / 1024, 256, 0, stream>>>(wv, wvb);
  to_bf16<<<wE / 1024, 256, 0, stream>>>(wo, wob);

  const dim3 gg(DIMSZ / 128, (NBATCH * S_LEN) / 128);  // (16, 32)
  gemm_nt_bf16<true><<<gg, 256, 0, stream>>>(xb, wqb, qb,  DIMSZ, DIMSZ);
  gemm_nt_bf16<true><<<gg, 256, 0, stream>>>(xb, wkb, kbf, DIMSZ, DIMSZ);
  gemm_nt_bf16<true><<<gg, 256, 0, stream>>>(xb, wvb, vbf, DIMSZ, DIMSZ);

  const int rope_threads = NBATCH * S_LEN * NHEAD * 64;
  rope_qk<<<rope_threads / 256, 256, 0, stream>>>(qb, kbf);

  flash_attn<<<dim3(S_LEN / 128, NHEAD, NBATCH), 256, 0, stream>>>(qb, kbf,
                                                                   vbf, abf);

  gemm_nt_bf16<false><<<gg, 256, 0, stream>>>(abf, wob, d_out, DIMSZ, DIMSZ);
}
